// MultiHeadAttention_34179349741880
// MI455X (gfx1250) — matrix operand order repaired and checked by reading, not compiled, not run on hardware
//
#include <hip/hip_runtime.h>

typedef __bf16 bf16_t;
typedef __attribute__((ext_vector_type(16))) __bf16 v16bf;
typedef __attribute__((ext_vector_type(8)))  __bf16 v8bf;
typedef __attribute__((ext_vector_type(8)))  float  v8f;

#define E_DIM   1024
#define HEADS   16
#define DHEAD   64
#define BATCH   2
#define SEQ     2048
#define MROWS   (SEQ * BATCH)
#define NEG_BIG -100000.0f

__device__ __forceinline__ bf16_t f2bf(float f) {
  union { float f; unsigned u; } v; v.f = f;
  unsigned r = (v.u + 0x7FFFu + ((v.u >> 16) & 1u)) >> 16;
  union { unsigned short s; bf16_t b; } o; o.s = (unsigned short)r;
  return o.b;
}

// ---------------------------------------------------------------------------
// Bulk fp32 -> bf16 (one pass; keeps conversion out of the GEMM inner loops)
// ---------------------------------------------------------------------------
__global__ void cvt_kernel(const float* __restrict__ X, bf16_t* __restrict__ Y) {
  const int idx = (blockIdx.x * blockDim.x + threadIdx.x) * 8;
  float4 a = *(const float4*)(X + idx);
  float4 b = *(const float4*)(X + idx + 4);
  v8bf o;
  o[0] = f2bf(a.x); o[1] = f2bf(a.y); o[2] = f2bf(a.z); o[3] = f2bf(a.w);
  o[4] = f2bf(b.x); o[5] = f2bf(b.y); o[6] = f2bf(b.z); o[7] = f2bf(b.w);
  *(v8bf*)(Y + idx) = o;
}

// ---------------------------------------------------------------------------
// Wt[k*E + j] = bf16(W[j*E + k])  (K-major weights -> contiguous B-fragments)
// ---------------------------------------------------------------------------
__global__ void wt_kernel(const float* __restrict__ W, bf16_t* __restrict__ Wt) {
  const int idx = blockIdx.x * blockDim.x + threadIdx.x;
  const int k = idx >> 10;
  const int j = idx & (E_DIM - 1);
  Wt[idx] = f2bf(W[j * E_DIM + k]);
}

// ---------------------------------------------------------------------------
// Y = X @ W^T + bias.  X is bf16 [MROWS, E].  Each wave: 32x64 tile,
// 2 A-fragments x 4 B-fragments -> 8 WMMAs per K-step, B reused 2x, A 4x.
// mode 0: bf16 out, (row, feature) layout      (Q, V)
// mode 1: bf16 out, transposed per (b,h): Kt[(ib*16+ih)*64+d][j]   (K)
// mode 2: f32 out + bias                        (final projection)
// ---------------------------------------------------------------------------
__global__ void proj_kernel(const bf16_t* __restrict__ X,
                            const bf16_t* __restrict__ Wt,
                            const float* __restrict__ bias,
                            bf16_t* __restrict__ outb,
                            float* __restrict__ outf,
                            int mode) {
  const int lane = threadIdx.x & 31;
  const int wave = threadIdx.x >> 5;
  const int lh = lane & 15;
  const int hw = lane >> 4;
  const int m0 = blockIdx.x * 32;
  const int n0 = blockIdx.y * 256 + wave * 64;

  v8f acc[2][4];
#pragma unroll
  for (int mt = 0; mt < 2; ++mt)
#pragma unroll
    for (int nt = 0; nt < 4; ++nt)
#pragma unroll
      for (int r = 0; r < 8; ++r) acc[mt][nt][r] = 0.0f;

  // A fragment: lane<16 holds row m0+lh, K = {0..7,16..23}; lanes>=16 offset 8.
  const bf16_t* ap = X + (size_t)(m0 + lh) * E_DIM + hw * 8;
  // B fragment: lane<16 holds column n0+lh, K = {0..7,16..23}; lanes>=16 offset 8.
  const bf16_t* bp = Wt + (size_t)(hw * 8) * E_DIM + n0 + lh;

  for (int k0 = 0; k0 < E_DIM; k0 += 32) {
    v16bf a[2];
#pragma unroll
    for (int mt = 0; mt < 2; ++mt) {
      v8bf lo = *(const v8bf*)(ap + (size_t)mt * 16 * E_DIM + k0);
      v8bf h8 = *(const v8bf*)(ap + (size_t)mt * 16 * E_DIM + k0 + 16);
#pragma unroll
      for (int i = 0; i < 8; ++i) { a[mt][i] = lo[i]; a[mt][8 + i] = h8[i]; }
    }
#pragma unroll
    for (int nt = 0; nt < 4; ++nt) {
      v16bf b;
#pragma unroll
      for (int i = 0; i < 8; ++i) { b[i] = bp[(size_t)(k0 + i) * E_DIM + nt * 16]; b[8 + i] = bp[(size_t)(k0 + 16 + i) * E_DIM + nt * 16]; }
      acc[0][nt] = __builtin_amdgcn_wmma_f32_16x16x32_bf16(false, a[0], false, b,
                                                           (short)0, acc[0][nt], false, false);
      acc[1][nt] = __builtin_amdgcn_wmma_f32_16x16x32_bf16(false, a[1], false, b,
                                                           (short)0, acc[1][nt], false, false);
    }
  }

#pragma unroll
  for (int mt = 0; mt < 2; ++mt) {
#pragma unroll
    for (int nt = 0; nt < 4; ++nt) {
      const int n = n0 + nt * 16 + lh;
      const float bn = bias[n];
      if (mode == 2) {
#pragma unroll
        for (int r = 0; r < 8; ++r)
          outf[(size_t)(m0 + mt * 16 + r + hw * 8) * E_DIM + n] = acc[mt][nt][r] + bn;
      } else if (mode == 0) {
#pragma unroll
        for (int r = 0; r < 8; ++r)
          outb[(size_t)(m0 + mt * 16 + r + hw * 8) * E_DIM + n] = f2bf(acc[mt][nt][r] + bn);
      } else {
        const int ihh = n >> 6, d = n & 63;
#pragma unroll
        for (int r = 0; r < 8; ++r) {
          const int row = m0 + mt * 16 + r + hw * 8;   // row = j*BATCH + ib
          const int ib = row & 1, j = row >> 1;
          outb[((size_t)(ib * HEADS + ihh) * DHEAD + d) * SEQ + j] = f2bf(acc[mt][nt][r] + bn);
        }
      }
    }
  }
}

// ---------------------------------------------------------------------------
// Flash attention, one wave per (batch*head, 16-query tile).
// Reference masks j < i (attends to j >= i), scale = 1/sqrt(E)=1/32.
// ---------------------------------------------------------------------------
__global__ void attn_kernel(const bf16_t* __restrict__ Q,
                            const bf16_t* __restrict__ Kt,
                            const bf16_t* __restrict__ V,
                            const unsigned char* __restrict__ in_mask,
                            bf16_t* __restrict__ pooled) {
  __shared__ bf16_t pbuf[16 * 32];
  const int lane = threadIdx.x & 31;
  const int lh = lane & 15;
  const int hw = lane >> 4;
  const int bh = blockIdx.x;           // 0..31
  const int ib = bh >> 4;
  const int ih = bh & 15;
  const int i0 = blockIdx.y * 16;

  // Q A-fragments for d-chunks {0..31},{32..63}
  v16bf qf[2];
  {
    const bf16_t* qp = Q + (size_t)((i0 + lh) * BATCH + ib) * E_DIM + ih * DHEAD + hw * 8;
#pragma unroll
    for (int t = 0; t < 2; ++t) {
      v8bf lo = *(const v8bf*)(qp + t * 32);
      v8bf h8 = *(const v8bf*)(qp + t * 32 + 16);
#pragma unroll
      for (int i = 0; i < 8; ++i) { qf[t][i] = lo[i]; qf[t][8 + i] = h8[i]; }
    }
  }

  const bf16_t* ktb = Kt + (size_t)bh * DHEAD * SEQ;   // [d][j], j contiguous
  const bf16_t* vb0 = V + (size_t)ib * E_DIM + ih * DHEAD;

  float rm[8], rl[8];
  v8f acc[4];
#pragma unroll
  for (int r = 0; r < 8; ++r) { rm[r] = -3.0e38f; rl[r] = 0.0f; }
#pragma unroll
  for (int t = 0; t < 4; ++t) {
#pragma unroll
    for (int r = 0; r < 8; ++r) acc[t][r] = 0.0f;
  }

  const float scale = 0.03125f;   // 1/sqrt(1024)

  for (int j0 = (i0 >> 5) << 5; j0 < SEQ; j0 += 32) {
    if (j0 + 32 < SEQ) {
      __builtin_prefetch((const void*)(ktb + (size_t)lane * SEQ + j0 + 32), 0, 1);
      __builtin_prefetch((const void*)(vb0 + (size_t)(j0 + 32 + lane) * (BATCH * E_DIM)), 0, 1);
    }
    // ---- scores S = Q K^T for 16 queries x 32 keys ----
    v16bf kb00, kb10, kb01, kb11;
    {
      const bf16_t* kp = ktb + (size_t)(hw * 8) * SEQ + j0 + lh;
#pragma unroll
      for (int i = 0; i < 8; ++i) {
        kb00[i] = kp[(size_t)i * SEQ];        kb00[8 + i] = kp[(size_t)(16 + i) * SEQ];
        kb10[i] = kp[(size_t)(32 + i) * SEQ]; kb10[8 + i] = kp[(size_t)(48 + i) * SEQ];
        kb01[i] = kp[(size_t)i * SEQ + 16];        kb01[8 + i] = kp[(size_t)(16 + i) * SEQ + 16];
        kb11[i] = kp[(size_t)(32 + i) * SEQ + 16]; kb11[8 + i] = kp[(size_t)(48 + i) * SEQ + 16];
      }
    }
    v8f s0 = {}, s1 = {};
    s0 = __builtin_amdgcn_wmma_f32_16x16x32_bf16(false, qf[0], false, kb00, (short)0, s0, false, false);
    s0 = __builtin_amdgcn_wmma_f32_16x16x32_bf16(false, qf[1], false, kb10, (short)0, s0, false, false);
    s1 = __builtin_amdgcn_wmma_f32_16x16x32_bf16(false, qf[0], false, kb01, (short)0, s1, false, false);
    s1 = __builtin_amdgcn_wmma_f32_16x16x32_bf16(false, qf[1], false, kb11, (short)0, s1, false, false);

    // ---- scale + masks (C layout: VGPR r -> row r / r+8, lane half = N) ----
    const int jA = j0 + lh, jB = j0 + 16 + lh;
    const bool mA = in_mask[jA * BATCH + ib] != 0;
    const bool mB = in_mask[jB * BATCH + ib] != 0;
#pragma unroll
    for (int r = 0; r < 8; ++r) {
      const int i = i0 + r + hw * 8;
      float a = s0[r] * scale; if (mA || jA < i) a = NEG_BIG;
      float b = s1[r] * scale; if (mB || jB < i) b = NEG_BIG;
      s0[r] = a; s1[r] = b;
    }

    // ---- online softmax: row stats across 16-lane half-groups ----
    float corr[8];
#pragma unroll
    for (int r = 0; r < 8; ++r) {
      float v = fmaxf(s0[r], s1[r]);
      v = fmaxf(v, __shfl_xor(v, 1, 32));
      v = fmaxf(v, __shfl_xor(v, 2, 32));
      v = fmaxf(v, __shfl_xor(v, 4, 32));
      v = fmaxf(v, __shfl_xor(v, 8, 32));
      const float m2 = fmaxf(rm[r], v);
      corr[r] = __expf(rm[r] - m2);
      rm[r] = m2;
    }
#pragma unroll
    for (int r = 0; r < 8; ++r) {
      const float p0 = __expf(s0[r] - rm[r]);
      const float p1 = __expf(s1[r] - rm[r]);
      float v = p0 + p1;
      v += __shfl_xor(v, 1, 32);
      v += __shfl_xor(v, 2, 32);
      v += __shfl_xor(v, 4, 32);
      v += __shfl_xor(v, 8, 32);
      rl[r] = rl[r] * corr[r] + v;
      const int M = r + hw * 8;
      pbuf[M * 32 + lh]      = f2bf(p0);   // C layout -> LDS
      pbuf[M * 32 + 16 + lh] = f2bf(p1);
    }
    __syncthreads();

    // ---- reload P in A-fragment layout ----
    v16bf pf;
    {
      const bf16_t* pp = pbuf + lh * 32 + hw * 8;
      v8bf lo = *(const v8bf*)(pp);
      v8bf h8 = *(const v8bf*)(pp + 16);
#pragma unroll
      for (int i = 0; i < 8; ++i) { pf[i] = lo[i]; pf[8 + i] = h8[i]; }
    }
    __syncthreads();

    // ---- O = O*corr + P @ V  (4 d-tiles of 16) ----
#pragma unroll
    for (int t = 0; t < 4; ++t) {
      v8f a2 = acc[t];
#pragma unroll
      for (int r = 0; r < 8; ++r) a2[r] *= corr[r];
      v16bf vfrag;
      {
        const bf16_t* vp = vb0 + (size_t)(j0 + hw * 8) * (BATCH * E_DIM) + t * 16 + lh;
#pragma unroll
        for (int i = 0; i < 8; ++i) { vfrag[i] = vp[(size_t)i * (BATCH * E_DIM)]; vfrag[8 + i] = vp[(size_t)(16 + i) * (BATCH * E_DIM)]; }
      }
      acc[t] = __builtin_amdgcn_wmma_f32_16x16x32_bf16(false, pf, false, vfrag,
                                                       (short)0, a2, false, false);
    }
  }

  // ---- normalize + store pooled (bf16, (m,b,e) layout) ----
#pragma unroll
  for (int t = 0; t < 4; ++t) {
#pragma unroll
    for (int r = 0; r < 8; ++r) {
      const int im = i0 + r + hw * 8;
      const int d = t * 16 + lh;
      pooled[(size_t)(im * BATCH + ib) * E_DIM + ih * DHEAD + d] = f2bf(acc[t][r] / rl[r]);
    }
  }
}

// ---------------------------------------------------------------------------
extern "C" void kernel_launch(void* const* d_in, const int* in_sizes, int n_in,
                              void* d_out, int out_size, void* d_ws, size_t ws_size,
                              hipStream_t stream) {
  (void)in_sizes; (void)n_in; (void)out_size; (void)ws_size;
  const float* queries = (const float*)d_in[0];
  const float* keys    = (const float*)d_in[1];
  const float* values  = (const float*)d_in[2];
  const unsigned char* in_mask = (const unsigned char*)d_in[3];
  const float* Wq = (const float*)d_in[4];
  const float* bq = (const float*)d_in[5];
  const float* Wk = (const float*)d_in[6];
  const float* bk = (const float*)d_in[7];
  const float* Wv = (const float*)d_in[8];
  const float* bv = (const float*)d_in[9];
  const float* Wo = (const float*)d_in[10];
  const float* bo = (const float*)d_in[11];
  float* out = (float*)d_out;

  char* ws = (char*)d_ws;
  const size_t MB = 1024ull * 1024ull;
  bf16_t* Wqt = (bf16_t*)(ws + 0 * MB);
  bf16_t* Wkt = (bf16_t*)(ws + 2 * MB);
  bf16_t* Wvt = (bf16_t*)(ws + 4 * MB);
  bf16_t* Wot = (bf16_t*)(ws + 6 * MB);
  bf16_t* Qb  = (bf16_t*)(ws + 8 * MB);    // 8 MB
  bf16_t* Ktb = (bf16_t*)(ws + 16 * MB);   // 8 MB
  bf16_t* Vb  = (bf16_t*)(ws + 24 * MB);   // 8 MB
  bf16_t* Xq  = (bf16_t*)(ws + 32 * MB);   // 8 MB (aliased by pooled later)
  bf16_t* Xk  = (bf16_t*)(ws + 40 * MB);   // 8 MB
  bf16_t* Xv  = (bf16_t*)(ws + 48 * MB);   // 8 MB   (total 56 MB)
  bf16_t* Pl  = Xq;                        // queries-bf16 dead after Q proj

  const int cvtBlocks = (MROWS * E_DIM) / (256 * 8);
  cvt_kernel<<<cvtBlocks, 256, 0, stream>>>(queries, Xq);
  cvt_kernel<<<cvtBlocks, 256, 0, stream>>>(keys,    Xk);
  cvt_kernel<<<cvtBlocks, 256, 0, stream>>>(values,  Xv);

  const int wtBlocks = (E_DIM * E_DIM) / 256;
  wt_kernel<<<wtBlocks, 256, 0, stream>>>(Wq, Wqt);
  wt_kernel<<<wtBlocks, 256, 0, stream>>>(Wk, Wkt);
  wt_kernel<<<wtBlocks, 256, 0, stream>>>(Wv, Wvt);
  wt_kernel<<<wtBlocks, 256, 0, stream>>>(Wo, Wot);

  dim3 pGrid(MROWS / 32, E_DIM / 256);
  proj_kernel<<<pGrid, 128, 0, stream>>>(Xq, Wqt, bq, Qb,  nullptr, 0);
  proj_kernel<<<pGrid, 128, 0, stream>>>(Xk, Wkt, bk, Ktb, nullptr, 1);
  proj_kernel<<<pGrid, 128, 0, stream>>>(Xv, Wvt, bv, Vb,  nullptr, 0);

  attn_kernel<<<dim3(BATCH * HEADS, SEQ / 16), 32, 0, stream>>>(Qb, Ktb, Vb, in_mask, Pl);

  proj_kernel<<<pGrid, 128, 0, stream>>>(Pl, Wot, bo, nullptr, out, 2);
}
